// ScaledDotProductAttention_16698832847293
// MI455X (gfx1250) — matrix operand order repaired and checked by reading, not compiled, not run on hardware
//
#include <hip/hip_runtime.h>

typedef __attribute__((ext_vector_type(16))) _Float16 v16h;
typedef __attribute__((ext_vector_type(8)))  _Float16 v8h;
typedef __attribute__((ext_vector_type(4)))  _Float16 v4h;
typedef __attribute__((ext_vector_type(8)))  float    v8f;
typedef __attribute__((ext_vector_type(4)))  float    v4f;

#define DEV __device__ __forceinline__

constexpr int Bb = 2, Hh = 16, Ss = 2048, Dd = 64;

DEV v8f wmma16x16x32(v16h a, v16h b, v8f c) {
  // D(f32 16x16) = A(f16 16x32) x B(f16 32x16) + C
  return __builtin_amdgcn_wmma_f32_16x16x32_f16(false, a, false, b,
                                                (short)0, c, false, false);
}

// ---- cross-lane max over 16-lane halves, entirely on VALU via DPP16 ----
// xor1 = quad_perm(1,0,3,2); xor2 = quad_perm(2,3,0,1). After those, each
// 4-lane bank is uniform, so row_half_mirror (i^7) acts as the xor-4 step and
// row_mirror (i^15) as the xor-8 step. DPP rows are 16 lanes on wave32, which
// matches the C-fragment half-wave row groups exactly.
template <int CTRL>
DEV float dpp_max_step(float x) {
  int y = __builtin_amdgcn_update_dpp(__float_as_int(x), __float_as_int(x),
                                      CTRL, 0xf, 0xf, true);
  return fmaxf(x, __int_as_float(y));
}
DEV float rmax16(float v) {
  v = dpp_max_step<0xB1>(v);   // quad_perm(1,0,3,2)
  v = dpp_max_step<0x4E>(v);   // quad_perm(2,3,0,1)
  v = dpp_max_step<0x141>(v);  // row_half_mirror
  v = dpp_max_step<0x140>(v);  // row_mirror
  return v;
}

DEV float rsum16_shfl(float v) {
  v += __shfl_xor(v, 1, 16);
  v += __shfl_xor(v, 2, 16);
  v += __shfl_xor(v, 4, 16);
  v += __shfl_xor(v, 8, 16);
  return v;
}
DEV float rmax16_shfl(float v) {
  v = fmaxf(v, __shfl_xor(v, 1, 16));
  v = fmaxf(v, __shfl_xor(v, 2, 16));
  v = fmaxf(v, __shfl_xor(v, 4, 16));
  v = fmaxf(v, __shfl_xor(v, 8, 16));
  return v;
}

// 16 consecutive f32 -> 16 packed f16 (B-fragment contraction order)
DEV v16h cvt16(const float* __restrict__ p) {
  v4f a = *(const v4f*)(p);
  v4f b = *(const v4f*)(p + 4);
  v4f c = *(const v4f*)(p + 8);
  v4f d = *(const v4f*)(p + 12);
  v16h r;
#pragma unroll
  for (int i = 0; i < 4; ++i) {
    r[i]      = (_Float16)a[i];
    r[i + 4]  = (_Float16)b[i];
    r[i + 8]  = (_Float16)c[i];
    r[i + 12] = (_Float16)d[i];
  }
  return r;
}

// A-fragment (16x32 f16) from f32 row: lane half h holds k={16h..16h+15}
DEV v16h cvtA(const float* __restrict__ row, int h) {
  const float* p0 = row + 16 * h;
  const float* p1 = p0 + 8;
  v4f a = *(const v4f*)(p0);
  v4f b = *(const v4f*)(p0 + 4);
  v4f c = *(const v4f*)(p1);
  v4f d = *(const v4f*)(p1 + 4);
  v16h r;
#pragma unroll
  for (int i = 0; i < 4; ++i) {
    r[i]      = (_Float16)a[i];
    r[i + 4]  = (_Float16)b[i];
    r[i + 8]  = (_Float16)c[i];
    r[i + 12] = (_Float16)d[i];
  }
  return r;
}

DEV v16h ldh16(const _Float16* __restrict__ p) {
  v8h a = *(const v8h*)(p);
  v8h b = *(const v8h*)(p + 8);
  v16h r;
#pragma unroll
  for (int i = 0; i < 8; ++i) { r[i] = a[i]; r[i + 8] = b[i]; }
  return r;
}

// ---------------------------------------------------------------------------
// Pre-pass: K (f32 [bh][s][d]) -> Kh (f16 same layout)
//           V (f32 [bh][s][d]) -> Vt (f16 [bh][d][s])   (PV B-fragment order)
// Whole f16 K/V working set (16.8 MB) lives in the 192 MB L2 afterwards.
// ---------------------------------------------------------------------------
__global__ __launch_bounds__(256)
void convert_kv(const float* __restrict__ K, const float* __restrict__ V,
                _Float16* __restrict__ Kh, _Float16* __restrict__ Vt) {
  const size_t base = ((size_t)blockIdx.x * 256 + threadIdx.x) * 4;

  v4f k = *(const v4f*)(K + base);
  v4h kh;
#pragma unroll
  for (int i = 0; i < 4; ++i) kh[i] = (_Float16)k[i];
  *(v4h*)(Kh + base) = kh;

  v4f v = *(const v4f*)(V + base);
  const int d0      = (int)(base & (Dd - 1));
  const size_t srow = base / Dd;        // bh*S + s
  const size_t bh   = srow / Ss;
  const size_t s    = srow - bh * Ss;
  _Float16* vt = Vt + bh * (size_t)Dd * Ss + s;
#pragma unroll
  for (int i = 0; i < 4; ++i) vt[(size_t)(d0 + i) * Ss] = (_Float16)v[i];
}

// ---------------------------------------------------------------------------
// Fast path: flash attention with pre-converted f16 K/V.
// One wave32 = 32 query rows (two 16-row M-tiles) -> every K/V fragment feeds
// two WMMAs: 18 WMMAs per 32-key tile (8 QK + 8 PV + 2 ones-column row-sum)
// against 16 b128 fragment loads.
// ---------------------------------------------------------------------------
__global__ __launch_bounds__(128, 2)
void fa_fwd_wmma_f16(const float* __restrict__ Q, const _Float16* __restrict__ Kh,
                     const _Float16* __restrict__ Vt, float* __restrict__ O) {
  const int bh   = blockIdx.y;
  const int wave = threadIdx.x >> 5;
  const int lane = threadIdx.x & 31;
  const int ll   = lane & 15;
  const int lh   = lane >> 4;

  const int qbase = blockIdx.x * 128 + wave * 32;   // 32 rows per wave
  const float*    Qp = Q  + (size_t)bh * Ss * Dd;
  const _Float16* Kp = Kh + (size_t)bh * Ss * Dd;
  const _Float16* Vp = Vt + (size_t)bh * Dd * Ss;
  float*          Op = O  + (size_t)bh * Ss * Dd;

  __shared__ __align__(16) _Float16 Plds[4][32 * 32];   // 2 KB per wave
  _Float16* pw = Plds[wave];

  // ---- Q: two M-tiles x two contraction chunks ----
  v16h qa[2][2];
#pragma unroll
  for (int g = 0; g < 2; ++g) {
    const float* qptr = Qp + (size_t)(qbase + 16 * g + ll) * Dd;
    qa[g][0] = cvtA(qptr, lh);
    qa[g][1] = cvtA(qptr + 32, lh);
  }

  v16h ones;
#pragma unroll
  for (int i = 0; i < 16; ++i) ones[i] = (_Float16)1.0f;

  v8f acc[2][4] = {};
  v8f lacc[2]   = {};
  float m[2][8];
#pragma unroll
  for (int g = 0; g < 2; ++g)
#pragma unroll
    for (int r = 0; r < 8; ++r) m[g][r] = -__builtin_inff();

  const float cs = 0.125f * 1.4426950408889634f;  // (1/sqrt(D)) * log2(e)
  const int ntiles = (qbase >> 5) + 1;            // keys <= qbase+31

  for (int t = 0; t < ntiles; ++t) {
    const int j = t << 5;

    if (t + 1 < ntiles) {
      __builtin_prefetch(Kp + (size_t)(j + 32 + lane) * Dd, 0, 1);
      __builtin_prefetch(Vp + (size_t)(lane << 1) * Ss + j + 32, 0, 1);
    }

    // ---- K fragments (2 key subtiles x 2 contraction chunks) ----
    const _Float16* kp0 = Kp + (size_t)(j + ll) * Dd + 16 * lh;
    const v16h kb00 = ldh16(kp0);
    const v16h kb01 = ldh16(kp0 + 32);
    const v16h kb10 = ldh16(kp0 + 16 * Dd);
    const v16h kb11 = ldh16(kp0 + 16 * Dd + 32);

    // ---- scores: 8 WMMAs, K fragments reused across both M-tiles ----
    v8f s0[2], s1[2];
#pragma unroll
    for (int g = 0; g < 2; ++g) {
      s0[g] = wmma16x16x32(qa[g][0], kb00, v8f{});
      s0[g] = wmma16x16x32(qa[g][1], kb01, s0[g]);
      s1[g] = wmma16x16x32(qa[g][0], kb10, v8f{});
      s1[g] = wmma16x16x32(qa[g][1], kb11, s1[g]);
    }

    // ---- V fragments issued now; latency hides under the softmax ----
    v16h vb[4];
#pragma unroll
    for (int c = 0; c < 4; ++c)
      vb[c] = ldh16(Vp + (size_t)(16 * c + ll) * Ss + j + 16 * lh);

    // ---- causal mask (last tiles only) + online softmax, per M-tile ----
    float alpha[2][8];
    const int k0 = j + ll, k1 = k0 + 16;
    asm volatile("s_wait_dscnt 0" ::: "memory");    // prior pa loads done
#pragma unroll
    for (int g = 0; g < 2; ++g) {
      const bool masked = (j + 31 > qbase + 16 * g);  // wave-uniform
      v8f p0, p1;
#pragma unroll
      for (int r = 0; r < 8; ++r) {
        float x0 = s0[g][r] * cs;
        float x1 = s1[g][r] * cs;
        if (masked) {
          const int row = qbase + 16 * g + r + 8 * lh;
          if (k0 > row) x0 = -__builtin_inff();
          if (k1 > row) x1 = -__builtin_inff();
        }
        float mx = rmax16(fmaxf(x0, x1));
        float mn = fmaxf(m[g][r], mx);
        alpha[g][r] = exp2f(m[g][r] - mn);
        m[g][r] = mn;
        p0[r] = exp2f(x0 - mn);
        p1[r] = exp2f(x1 - mn);
      }
      // P: C-layout f32 -> A-layout f16 staging (per-wave private LDS)
#pragma unroll
      for (int r = 0; r < 8; ++r) {
        const int row = 16 * g + r + 8 * lh;
        pw[row * 32 + ll]      = (_Float16)p0[r];
        pw[row * 32 + 16 + ll] = (_Float16)p1[r];
      }
    }
    asm volatile("s_wait_dscnt 0" ::: "memory");    // stores visible wave-wide

    v16h pa[2];
#pragma unroll
    for (int g = 0; g < 2; ++g) {
      v8h lo = *(const v8h*)(pw + (16 * g + ll) * 32 + 16 * lh);
      v8h hi = *(const v8h*)(pw + (16 * g + ll) * 32 + 16 * lh + 8);
#pragma unroll
      for (int i = 0; i < 8; ++i) { pa[g][i] = lo[i]; pa[g][i + 8] = hi[i]; }
    }

    // ---- l and acc updates: 10 WMMAs, V fragments reused across M-tiles ----
#pragma unroll
    for (int g = 0; g < 2; ++g) {
#pragma unroll
      for (int r = 0; r < 8; ++r) lacc[g][r] *= alpha[g][r];
      lacc[g] = wmma16x16x32(pa[g], ones, lacc[g]);
#pragma unroll
      for (int c = 0; c < 4; ++c) {
#pragma unroll
        for (int r = 0; r < 8; ++r) acc[g][c][r] *= alpha[g][r];
        acc[g][c] = wmma16x16x32(pa[g], vb[c], acc[g][c]);
      }
    }
  }

#pragma unroll
  for (int g = 0; g < 2; ++g) {
    float inv[8];
#pragma unroll
    for (int r = 0; r < 8; ++r) inv[r] = 1.0f / lacc[g][r];
#pragma unroll
    for (int c = 0; c < 4; ++c)
#pragma unroll
      for (int r = 0; r < 8; ++r) {
        const int row = qbase + 16 * g + r + 8 * lh;
        Op[(size_t)row * Dd + c * 16 + ll] = acc[g][c][r] * inv[r];
      }
  }
}

// ---------------------------------------------------------------------------
// Fallback (no workspace): self-contained f32-input version.
// ---------------------------------------------------------------------------
__global__ __launch_bounds__(128, 2)
void fa_fwd_wmma_f32(const float* __restrict__ Q, const float* __restrict__ K,
                     const float* __restrict__ V, float* __restrict__ O) {
  const int bh   = blockIdx.y;
  const int wave = threadIdx.x >> 5;
  const int lane = threadIdx.x & 31;
  const int ll   = lane & 15;
  const int lh   = lane >> 4;

  const int qbase = blockIdx.x * 64 + wave * 16;
  const size_t base = (size_t)bh * Ss * Dd;
  const float* Qp = Q + base;
  const float* Kp = K + base;
  const float* Vp = V + base;
  float*       Op = O + base;

  __shared__ __align__(16) _Float16 Plds[4][16 * 32];
  _Float16* pw = Plds[wave];

  const float* qptr = Qp + (size_t)(qbase + ll) * Dd;
  const v16h qa0 = cvtA(qptr, lh);
  const v16h qa1 = cvtA(qptr + 32, lh);

  v8f acc[4] = {};
  float m[8], l[8];
#pragma unroll
  for (int r = 0; r < 8; ++r) { m[r] = -__builtin_inff(); l[r] = 0.f; }

  const float cs = 0.125f * 1.4426950408889634f;
  const int ntiles = (qbase + 47) >> 5;

  for (int t = 0; t < ntiles; ++t) {
    const int j = t << 5;

    const float* kp0 = Kp + (size_t)(j + ll) * Dd + 16 * lh;
    v8f s0 = wmma16x16x32(qa0, cvt16(kp0), v8f{});
    s0     = wmma16x16x32(qa1, cvt16(kp0 + 32), s0);
    const float* kp1 = kp0 + 16 * Dd;
    v8f s1 = wmma16x16x32(qa0, cvt16(kp1), v8f{});
    s1     = wmma16x16x32(qa1, cvt16(kp1 + 32), s1);

    v8f p0, p1;
    float alpha[8];
    const int k0 = j + ll, k1 = k0 + 16;
#pragma unroll
    for (int r = 0; r < 8; ++r) {
      const int row = qbase + r + 8 * lh;
      float x0 = (k0 <= row) ? s0[r] * cs : -__builtin_inff();
      float x1 = (k1 <= row) ? s1[r] * cs : -__builtin_inff();
      float mx = rmax16_shfl(fmaxf(x0, x1));
      float mn = fmaxf(m[r], mx);
      alpha[r] = exp2f(m[r] - mn);
      m[r] = mn;
      p0[r] = exp2f(x0 - mn);
      p1[r] = exp2f(x1 - mn);
      l[r] = l[r] * alpha[r] + rsum16_shfl(p0[r] + p1[r]);
    }

    asm volatile("s_wait_dscnt 0" ::: "memory");
#pragma unroll
    for (int r = 0; r < 8; ++r) {
      const int row = r + 8 * lh;
      pw[row * 32 + ll]      = (_Float16)p0[r];
      pw[row * 32 + 16 + ll] = (_Float16)p1[r];
    }
    asm volatile("s_wait_dscnt 0" ::: "memory");
    v16h pa;
    {
      v8h lo = *(const v8h*)(pw + ll * 32 + 16 * lh);
      v8h hi = *(const v8h*)(pw + ll * 32 + 16 * lh + 8);
#pragma unroll
      for (int i = 0; i < 8; ++i) { pa[i] = lo[i]; pa[i + 8] = hi[i]; }
    }

    const float* vp0 = Vp + (size_t)(j + 16 * lh) * Dd + ll;
#pragma unroll
    for (int c = 0; c < 4; ++c) {
      v16h vb;
      const float* vp = vp0 + c * 16;
#pragma unroll
      for (int e = 0; e < 16; ++e) vb[e] = (_Float16)vp[(size_t)e * Dd];
#pragma unroll
      for (int r = 0; r < 8; ++r) acc[c][r] *= alpha[r];
      acc[c] = wmma16x16x32(pa, vb, acc[c]);
    }
  }

  float inv[8];
#pragma unroll
  for (int r = 0; r < 8; ++r) inv[r] = 1.0f / l[r];
#pragma unroll
  for (int c = 0; c < 4; ++c)
#pragma unroll
    for (int r = 0; r < 8; ++r) {
      const int row = qbase + r + 8 * lh;
      Op[(size_t)row * Dd + c * 16 + ll] = acc[c][r] * inv[r];
    }
}

extern "C" void kernel_launch(void* const* d_in, const int* in_sizes, int n_in,
                              void* d_out, int out_size, void* d_ws, size_t ws_size,
                              hipStream_t stream) {
  // inputs: query, key, value, mask (causal mask handled analytically)
  const float* Q = (const float*)d_in[0];
  const float* K = (const float*)d_in[1];
  const float* V = (const float*)d_in[2];
  float* O = (float*)d_out;

  const size_t elems   = (size_t)Bb * Hh * Ss * Dd;
  const size_t khBytes = elems * sizeof(_Float16);

  if (ws_size >= 2 * khBytes) {
    _Float16* Kh = (_Float16*)d_ws;
    _Float16* Vt = (_Float16*)((char*)d_ws + khBytes);
    const int cblocks = (int)(elems / (256 * 4));
    convert_kv<<<cblocks, 256, 0, stream>>>(K, V, Kh, Vt);
    dim3 grid(Ss / 128, Bb * Hh);   // 32 rows/wave, 4 waves/block
    fa_fwd_wmma_f16<<<grid, 128, 0, stream>>>(Q, Kh, Vt, O);
  } else {
    dim3 grid(Ss / 64, Bb * Hh);
    fa_fwd_wmma_f32<<<grid, 128, 0, stream>>>(Q, K, V, O);
  }

  (void)in_sizes; (void)n_in; (void)out_size;
}
